// TorchAttention_89644557402990
// MI455X (gfx1250) — matrix operand order repaired and checked by reading, not compiled, not run on hardware
//
#include <hip/hip_runtime.h>
#include <math.h>

typedef _Float16 h16v __attribute__((ext_vector_type(16)));
typedef _Float16 h8v  __attribute__((ext_vector_type(8)));
typedef _Float16 h4v  __attribute__((ext_vector_type(4)));
typedef float    f8v  __attribute__((ext_vector_type(8)));
typedef float    f4v  __attribute__((ext_vector_type(4)));

namespace {
constexpr int   kS = 2048;
constexpr int   kH = 24;
constexpr int   kD = 128;
// 1/sqrt(128) * log2(e): softmax computed in base-2 domain -> exp2 is a bare v_exp_f32
constexpr float kScaleL2 = 0.08838834764831845f * 1.4426950408889634f;
constexpr float kNegBig  = -3.0e30f;
}

// Flash attention fwd, causal, fp32 I/O, f16 WMMA math, fp32 softmax/accum.
// One workgroup = 4 wave32 = 64 query rows of one head.
// Double-buffered LDS staging: global loads for tile t+1 issue before the
// compute of tile t; convert+LDS-store happen after compute; 1 barrier/iter.
__global__ __launch_bounds__(128, 1) void fa_fwd(const float* __restrict__ Qg,
                                                 const float* __restrict__ Kg,
                                                 const float* __restrict__ Vg,
                                                 float* __restrict__ Og) {
  __shared__ __align__(32) _Float16 Kt[2][32][128];   // K tiles, row-major f16
  __shared__ __align__(32) _Float16 Vt[2][128][32];   // V tiles, transposed f16
  __shared__ __align__(32) _Float16 Pb[4][16][32];    // per-wave P staging

  const int tid  = threadIdx.x;
  const int lane = tid & 31;
  const int wave = tid >> 5;
  const int head = blockIdx.x % kH;
  const int qt   = blockIdx.x / kH;
  const int qbase = qt * 64;

  const int  n      = lane & 15;      // A: row M ; B: column N
  const bool hi     = lane >= 16;
  const int  kbA    = hi ? 16 : 0;    // A-operand K base (contiguous k = 16h + s)
  const int  kbB    = hi ? 16 : 0;    // B-operand K base
  const int  rowoff = hi ? 8 : 0;     // C-layout row offset

  // staging geometry for this thread (8 float4 per tile per matrix)
  const int st_row = tid >> 2;              // unused; the staging row is i>>5 (see below)
  // i = tid + it*128 -> row = i>>5, c4 = (i&31)<<2 ; precompute per-it below.

  // ---- load this wave's Q rows once, fp32 -> f16, into 4 A operands ----
  const int qrowA = qbase + wave * 16 + n;
  const float* qptr = Qg + ((size_t)qrowA * kH + head) * kD;
  h16v qa[4];
#pragma unroll
  for (int ks = 0; ks < 4; ++ks) {
    const float* p = qptr + ks * 32 + kbA;
    f4v c0 = *(const f4v*)(p);
    f4v c1 = *(const f4v*)(p + 4);
    f4v c2 = *(const f4v*)(p + 8);
    f4v c3 = *(const f4v*)(p + 12);
#pragma unroll
    for (int e = 0; e < 4; ++e) {
      qa[ks][e]      = (_Float16)c0[e];
      qa[ks][4 + e]  = (_Float16)c1[e];
      qa[ks][8 + e]  = (_Float16)c2[e];
      qa[ks][12 + e] = (_Float16)c3[e];
    }
  }

  const f8v fzero = {};
  f8v oacc[8];
#pragma unroll
  for (int c = 0; c < 8; ++c) oacc[c] = fzero;
  float rmax[8], rsum[8];
#pragma unroll
  for (int r = 0; r < 8; ++r) { rmax[r] = kNegBig; rsum[r] = 0.0f; }

  const int ntiles = (qbase + 64) >> 5;   // causal: tiles up to the diagonal
  (void)st_row;

  f4v pk[8], pv[8];                        // raw-f32 prefetch registers

  // ---- prologue: fetch tile 0 and stage into LDS buffer 0 ----
#pragma unroll
  for (int it = 0; it < 8; ++it) {
    const int i   = tid + it * 128;
    const int row = i >> 5;
    const int c4  = (i & 31) << 2;
    const size_t goff = ((size_t)row * kH + head) * kD + c4;
    pk[it] = *(const f4v*)(Kg + goff);
    pv[it] = *(const f4v*)(Vg + goff);
  }
#pragma unroll
  for (int it = 0; it < 8; ++it) {
    const int i   = tid + it * 128;
    const int row = i >> 5;
    const int c4  = (i & 31) << 2;
    h4v kh;
#pragma unroll
    for (int j = 0; j < 4; ++j) kh[j] = (_Float16)pk[it][j];
    *(h4v*)&Kt[0][row][c4] = kh;
#pragma unroll
    for (int j = 0; j < 4; ++j) Vt[0][c4 + j][row] = (_Float16)pv[it][j];
  }

  for (int t = 0; t < ntiles; ++t) {
    const int kvb = t * 32;
    const int cur = t & 1;
    const bool pf = (t + 1) < ntiles;

    __syncthreads();   // staging of buffer `cur` complete; prev reads done

    // ---- issue global loads for tile t+1 (land during this tile's math) ----
    if (pf) {
      const int kvn = kvb + 32;
#pragma unroll
      for (int it = 0; it < 8; ++it) {
        const int i   = tid + it * 128;
        const int row = i >> 5;
        const int c4  = (i & 31) << 2;
        const size_t goff = ((size_t)(kvn + row) * kH + head) * kD + c4;
        pk[it] = *(const f4v*)(Kg + goff);
        pv[it] = *(const f4v*)(Vg + goff);
      }
    }

    // ---- scores S = Q * K^T (two 16x16 blocks; preload all 4 B-operands) ----
    f8v sc[2];
#pragma unroll
    for (int nh = 0; nh < 2; ++nh) {
      const _Float16* kr = &Kt[cur][nh * 16 + n][kbB];
      h16v b0 = *(const h16v*)(kr);
      h16v b1 = *(const h16v*)(kr + 32);
      h16v b2 = *(const h16v*)(kr + 64);
      h16v b3 = *(const h16v*)(kr + 96);
      f8v acc = fzero;
      acc = __builtin_amdgcn_wmma_f32_16x16x32_f16(false, qa[0], false, b0, (short)0, acc, false, false);
      acc = __builtin_amdgcn_wmma_f32_16x16x32_f16(false, qa[1], false, b1, (short)0, acc, false, false);
      acc = __builtin_amdgcn_wmma_f32_16x16x32_f16(false, qa[2], false, b2, (short)0, acc, false, false);
      acc = __builtin_amdgcn_wmma_f32_16x16x32_f16(false, qa[3], false, b3, (short)0, acc, false, false);
      sc[nh] = acc;
    }

    // ---- causal mask (only near the diagonal) + online softmax (base-2) ----
    const bool needs_mask = (kvb + 31) > (qbase + wave * 16);  // wave-uniform
#pragma unroll
    for (int r = 0; r < 8; ++r) {
      float s0 = sc[0][r] * kScaleL2;
      float s1 = sc[1][r] * kScaleL2;
      if (needs_mask) {
        const int qr = qbase + wave * 16 + r + rowoff;
        s0 = ((kvb + n)      <= qr) ? s0 : kNegBig;
        s1 = ((kvb + 16 + n) <= qr) ? s1 : kNegBig;
      }

      float mx = fmaxf(s0, s1);
#pragma unroll
      for (int off = 8; off >= 1; off >>= 1)
        mx = fmaxf(mx, __shfl_xor(mx, off, 32));
      const float nm   = fmaxf(rmax[r], mx);
      const float corr = __builtin_amdgcn_exp2f(rmax[r] - nm);
      rmax[r] = nm;

      const float p0 = __builtin_amdgcn_exp2f(s0 - nm);
      const float p1 = __builtin_amdgcn_exp2f(s1 - nm);
      float rs = p0 + p1;
#pragma unroll
      for (int off = 8; off >= 1; off >>= 1)
        rs += __shfl_xor(rs, off, 32);
      rsum[r] = rsum[r] * corr + rs;
#pragma unroll
      for (int c = 0; c < 8; ++c) oacc[c][r] *= corr;

      // stage P (f16) for layout change C -> A (per-wave buffer)
      Pb[wave][r + rowoff][n]      = (_Float16)p0;
      Pb[wave][r + rowoff][16 + n] = (_Float16)p1;
    }
    __builtin_amdgcn_wave_barrier();
    asm volatile("" ::: "memory");  // same-wave LDS ops are in order (ISA §7.1)

    // ---- O += P * V  (K = 32 = kv tile; preload B-operands in groups of 4) ----
    {
      const _Float16* pr = &Pb[wave][n][0];
      h8v pa0 = *(const h8v*)(pr + kbA);
      h8v pa1 = *(const h8v*)(pr + kbA + 8);
      h16v pa = __builtin_shufflevector(pa0, pa1,
                                        0, 1, 2, 3, 4, 5, 6, 7,
                                        8, 9, 10, 11, 12, 13, 14, 15);
#pragma unroll
      for (int g = 0; g < 2; ++g) {
        const int cb = g * 4;
        h16v v0 = *(const h16v*)(&Vt[cur][(cb + 0) * 16 + n][kbB]);
        h16v v1 = *(const h16v*)(&Vt[cur][(cb + 1) * 16 + n][kbB]);
        h16v v2 = *(const h16v*)(&Vt[cur][(cb + 2) * 16 + n][kbB]);
        h16v v3 = *(const h16v*)(&Vt[cur][(cb + 3) * 16 + n][kbB]);
        oacc[cb + 0] = __builtin_amdgcn_wmma_f32_16x16x32_f16(false, pa, false, v0, (short)0, oacc[cb + 0], false, false);
        oacc[cb + 1] = __builtin_amdgcn_wmma_f32_16x16x32_f16(false, pa, false, v1, (short)0, oacc[cb + 1], false, false);
        oacc[cb + 2] = __builtin_amdgcn_wmma_f32_16x16x32_f16(false, pa, false, v2, (short)0, oacc[cb + 2], false, false);
        oacc[cb + 3] = __builtin_amdgcn_wmma_f32_16x16x32_f16(false, pa, false, v3, (short)0, oacc[cb + 3], false, false);
      }
    }

    // ---- stage tile t+1 into the other LDS buffer (read next iter) ----
    if (pf) {
      const int nxt = cur ^ 1;
#pragma unroll
      for (int it = 0; it < 8; ++it) {
        const int i   = tid + it * 128;
        const int row = i >> 5;
        const int c4  = (i & 31) << 2;
        h4v kh;
#pragma unroll
        for (int j = 0; j < 4; ++j) kh[j] = (_Float16)pk[it][j];
        *(h4v*)&Kt[nxt][row][c4] = kh;
#pragma unroll
        for (int j = 0; j < 4; ++j) Vt[nxt][c4 + j][row] = (_Float16)pv[it][j];
      }
    }
  }

  // ---- normalize and write out [S, H*D] ----
#pragma unroll
  for (int r = 0; r < 8; ++r) {
    const float inv = 1.0f / rsum[r];
#pragma unroll
    for (int c = 0; c < 8; ++c) oacc[c][r] *= inv;
  }
#pragma unroll
  for (int c = 0; c < 8; ++c) {
#pragma unroll
    for (int r = 0; r < 8; ++r) {
      const int qr = qbase + wave * 16 + r + rowoff;
      Og[(size_t)qr * (kH * kD) + head * kD + c * 16 + n] = oacc[c][r];
    }
  }
}

extern "C" void kernel_launch(void* const* d_in, const int* in_sizes, int n_in,
                              void* d_out, int out_size, void* d_ws, size_t ws_size,
                              hipStream_t stream) {
  (void)in_sizes; (void)n_in; (void)out_size; (void)d_ws; (void)ws_size;
  const float* Q = (const float*)d_in[0];
  const float* K = (const float*)d_in[1];
  const float* V = (const float*)d_in[2];
  float* O = (float*)d_out;
  dim3 grid((kS / 64) * kH); // 32 query tiles * 24 heads = 768 workgroups
  dim3 block(128);           // 4 wave32
  fa_fwd<<<grid, block, 0, stream>>>(Q, K, V, O);
}
